// AttnBlock_30150670418470
// MI455X (gfx1250) — compile-verified, matrix operand order checked by reading, not run on hardware
//
#include <hip/hip_runtime.h>
#include <hip/hip_bf16.h>
#include <math.h>

#define BATCH  2
#define CDIM   512
#define NPIX   4096
#define GROUPS 32
#define CPG    16
#define EPSV   1e-6f

typedef __attribute__((ext_vector_type(16))) _Float16 v16h;
typedef __attribute__((ext_vector_type(8)))  _Float16 v8h;
typedef __attribute__((ext_vector_type(8)))  float    v8f;

// Load a 16x32 f16 WMMA fragment (A-style; also valid for B when the source is
// stored so the B-column is contiguous along K). Source is row-major [row][k]
// with leading dim `ld` (halves). Per ISA 7.12.2: lanes 0-15 hold K chunks
// {k0..k0+7, k0+16..k0+23}; lanes 16-31 hold {k0+8..k0+15, k0+24..k0+31}.
static __device__ __forceinline__ v16h load_frag(const _Float16* __restrict__ base,
                                                 int row0, int ld, int k0, int lane) {
  int r  = row0 + (lane & 15);
  int kk = k0 + ((lane >> 4) << 3);
  const _Float16* p = base + (size_t)r * ld + kk;
  v8h lo = *(const v8h*)(p);
  v8h hi = *(const v8h*)(p + 16);
  v16h f;
#pragma unroll
  for (int i = 0; i < 8; ++i) { f[i] = lo[i]; f[i + 8] = hi[i]; }
  return f;
}

static __device__ __forceinline__ v8f wmma_f16(v16h a, v16h b, v8f c) {
  return __builtin_amdgcn_wmma_f32_16x16x32_f16(false, a, false, b, (short)0, c, false, false);
}

// ---------------- GroupNorm statistics: one block per (batch, group) ----------
__global__ void gn_stats_kernel(const float* __restrict__ x, float* __restrict__ stats) {
  int bg = blockIdx.x;  // 0..63 ; group data is a contiguous CPG*NPIX chunk
  const float* p = x + (size_t)bg * CPG * NPIX;
  float s = 0.f, ss = 0.f;
  for (int t = threadIdx.x; t < CPG * NPIX; t += 256) {
    float v = p[t]; s += v; ss += v * v;
  }
  __shared__ float rs[256], rss[256];
  rs[threadIdx.x] = s; rss[threadIdx.x] = ss;
  __syncthreads();
  for (int o = 128; o > 0; o >>= 1) {
    if (threadIdx.x < o) { rs[threadIdx.x] += rs[threadIdx.x + o]; rss[threadIdx.x] += rss[threadIdx.x + o]; }
    __syncthreads();
  }
  if (threadIdx.x == 0) {
    const float inv = 1.0f / (float)(CPG * NPIX);
    float mean = rs[0] * inv;
    float var  = rss[0] * inv - mean * mean;
    stats[bg * 2]     = mean;
    stats[bg * 2 + 1] = rsqrtf(var + EPSV);
  }
}

// ---------------- Weight f32 -> f16 conversion --------------------------------
__global__ void w2h_kernel(const float* __restrict__ wq, const float* __restrict__ wk,
                           const float* __restrict__ wv, const float* __restrict__ wo,
                           _Float16* __restrict__ hq, _Float16* __restrict__ hk,
                           _Float16* __restrict__ hv, _Float16* __restrict__ ho) {
  int i = blockIdx.x * 256 + threadIdx.x;
  if (i < CDIM * CDIM) {
    hq[i] = (_Float16)wq[i];
    hk[i] = (_Float16)wk[i];
    hv[i] = (_Float16)wv[i];
    ho[i] = (_Float16)wo[i];
  }
}

// ---------------- GroupNorm apply -> hnT[b][n][c] (f16, c contiguous) ---------
__global__ void gn_apply_kernel(const float* __restrict__ x, const float* __restrict__ stats,
                                const float* __restrict__ gnw, const float* __restrict__ gnb,
                                _Float16* __restrict__ hnT) {
  int idx = blockIdx.x * 256 + threadIdx.x;          // [b][c][n], n fastest
  int n = idx & (NPIX - 1);
  int c = (idx >> 12) & (CDIM - 1);
  int b = idx >> 21;
  int g = c >> 4;
  float mean = stats[(b * GROUPS + g) * 2];
  float rstd = stats[(b * GROUPS + g) * 2 + 1];
  float v = (x[idx] - mean) * rstd * gnw[c] + gnb[c];
  hnT[((size_t)b * NPIX + n) * CDIM + c] = (_Float16)v;
}

// ---------------- Fused Q/K/V projection GEMM ---------------------------------
// grid (NPIX/32, CDIM/64, BATCH), 8 waves: 4x2 tiling of a 64(m) x 32(n) tile.
// The attention softmax scale (C^-0.5) is folded into the Q output here.
__global__ void qkv_kernel(const _Float16* __restrict__ hnT,
                           const _Float16* __restrict__ wq, const _Float16* __restrict__ wk,
                           const _Float16* __restrict__ wv,
                           const float* __restrict__ bq, const float* __restrict__ bk,
                           const float* __restrict__ bv,
                           _Float16* __restrict__ qT, _Float16* __restrict__ kT,
                           _Float16* __restrict__ vM) {
  int lane = threadIdx.x & 31;
  int w    = threadIdx.x >> 5;
  int m0 = blockIdx.y * 64 + (w & 3) * 16;
  int n0 = blockIdx.x * 32 + (w >> 2) * 16;
  int b  = blockIdx.z;
  const _Float16* hb = hnT + (size_t)b * NPIX * CDIM;
  const float scale = 0.044194173824159216f;  // 512^-0.5

  v8f aq = {}, ak = {}, av = {};
  for (int k0 = 0; k0 < CDIM; k0 += 32) {
    v16h bf = load_frag(hb, n0, CDIM, k0, lane);   // B: K=c, N=pixel
    v16h fa = load_frag(wq, m0, CDIM, k0, lane);
    aq = wmma_f16(fa, bf, aq);
    fa = load_frag(wk, m0, CDIM, k0, lane);
    ak = wmma_f16(fa, bf, ak);
    fa = load_frag(wv, m0, CDIM, k0, lane);
    av = wmma_f16(fa, bf, av);
  }
  int n  = n0 + (lane & 15);
  int mb = m0 + ((lane >> 4) << 3);
  size_t obase = (size_t)b * NPIX * CDIM;
  v8h oq, ok;
#pragma unroll
  for (int e = 0; e < 8; ++e) {
    oq[e] = (_Float16)((aq[e] + bq[mb + e]) * scale);   // pre-scaled Q
    ok[e] = (_Float16)(ak[e] + bk[mb + e]);
  }
  *(v8h*)(qT + obase + (size_t)n * CDIM + mb) = oq;   // [n][c]
  *(v8h*)(kT + obase + (size_t)n * CDIM + mb) = ok;   // [n][c]
#pragma unroll
  for (int e = 0; e < 8; ++e)                          // V kept [c][n]
    vM[obase + (size_t)(mb + e) * NPIX + n] = (_Float16)(av[e] + bv[mb + e]);
}

// ---------------- Flash attention: block = 32 query rows, 8 waves -------------
// S phase: wave (is = w>>2, jc = w&3) computes two 16x16 score tiles of a
//          32(i) x 128(j) block.
// PV phase: wave (is2 = w&1, cc = w>>1) owns a 16(i) x 128(c) slice of O.
__global__ void attn_kernel(const _Float16* __restrict__ qT, const _Float16* __restrict__ kT,
                            const _Float16* __restrict__ vM, _Float16* __restrict__ oT) {
  __shared__ _Float16 Qs[32 * CDIM];   // 32 KB, raw copy of 32 Q rows
  __shared__ _Float16 Ps[32 * 128];    // 8 KB probabilities for current j-block
  __shared__ float wmax[4][32], wsum[4][32];
  __shared__ float rowm[32], rowl[32], newm[32], resc[32];

  int lane = threadIdx.x & 31;
  int w    = threadIdx.x >> 5;
  int b    = blockIdx.y;
  int i0   = blockIdx.x * 32;
  size_t base = (size_t)b * NPIX * CDIM;
  const _Float16* qb = qT + base;
  const _Float16* kb = kT + base;
  const _Float16* vb = vM + base;

  // ---- async DMA of the 32 KB Q block straight into LDS (no VGPR round-trip)
  {
    unsigned lds0 = (unsigned)(size_t)(&Qs[0]);
    const char* g0 = (const char*)(qb + (size_t)i0 * CDIM);
    for (int t = threadIdx.x; t < 2048; t += 256) {   // 2048 x 16B = 32 KB
      unsigned l = lds0 + t * 16;
      const char* g = g0 + (size_t)t * 16;
      asm volatile("global_load_async_to_lds_b128 %0, %1, off"
                   :: "v"(l), "v"(g) : "memory");
    }
  }
  if (threadIdx.x < 32) { rowm[threadIdx.x] = -3.0e38f; rowl[threadIdx.x] = 0.f; }
  asm volatile("s_wait_asynccnt 0x0" ::: "memory");
  __syncthreads();

  v8f acc[8] = {{}, {}, {}, {}, {}, {}, {}, {}};  // wave O slice: 16 i x 128 c
  int is  = w >> 2, jc = w & 3;    // S-phase role
  int is2 = w & 1,  cc = w >> 1;   // PV-phase role
  int mb = ((lane >> 4) << 3);

  for (int j0 = 0; j0 < NPIX; j0 += 128) {
    // ---- S sub-tiles (16 x 32) for this wave at columns j0 + 32*jc
    int jw0 = j0 + 32 * jc;
    if (j0 + 128 < NPIX)  // hint next j-block's K rows into cache
      __builtin_prefetch(kb + (size_t)(jw0 + 128 + (lane & 15)) * CDIM, 0, 1);
    v8f s0 = {}, s1 = {};
    for (int k0 = 0; k0 < CDIM; k0 += 32) {
      v16h fa  = load_frag(Qs + is * 16 * CDIM, 0, CDIM, k0, lane);  // A: rows=i
      v16h fb0 = load_frag(kb, jw0,      CDIM, k0, lane);            // B: K=c,N=j
      v16h fb1 = load_frag(kb, jw0 + 16, CDIM, k0, lane);
      s0 = wmma_f16(fa, fb0, s0);
      s1 = wmma_f16(fa, fb1, s1);
    }
    // ---- per-row max over this wave's 32 columns (half-wave reduction)
    float rmax[8];
#pragma unroll
    for (int e = 0; e < 8; ++e) {
      float vv = fmaxf(s0[e], s1[e]);
#pragma unroll
      for (int off = 1; off < 16; off <<= 1) vv = fmaxf(vv, __shfl_xor(vv, off, 32));
      rmax[e] = vv;
    }
    if ((lane & 15) == 0) {
#pragma unroll
      for (int e = 0; e < 8; ++e) wmax[jc][is * 16 + mb + e] = rmax[e];
    }
    __syncthreads();
    if (threadIdx.x < 32) {
      float nm = rowm[threadIdx.x];
      for (int jj = 0; jj < 4; ++jj) nm = fmaxf(nm, wmax[jj][threadIdx.x]);
      newm[threadIdx.x] = nm;
      resc[threadIdx.x] = __expf(rowm[threadIdx.x] - nm);
      rowm[threadIdx.x] = nm;
    }
    __syncthreads();
    // ---- P = exp(S - m), local row sums, stash P in LDS as f16
    float rsum[8];
#pragma unroll
    for (int e = 0; e < 8; ++e) {
      int r = is * 16 + mb + e;
      float m = newm[r];
      float p0 = __expf(s0[e] - m);
      float p1 = __expf(s1[e] - m);
      float sv = p0 + p1;
#pragma unroll
      for (int off = 1; off < 16; off <<= 1) sv += __shfl_xor(sv, off, 32);
      rsum[e] = sv;
      Ps[r * 128 + 32 * jc + (lane & 15)]      = (_Float16)p0;
      Ps[r * 128 + 32 * jc + 16 + (lane & 15)] = (_Float16)p1;
    }
    if ((lane & 15) == 0) {
#pragma unroll
      for (int e = 0; e < 8; ++e) wsum[jc][is * 16 + mb + e] = rsum[e];
    }
    __syncthreads();
    if (threadIdx.x < 32) {
      float l = rowl[threadIdx.x] * resc[threadIdx.x];
      for (int jj = 0; jj < 4; ++jj) l += wsum[jj][threadIdx.x];
      rowl[threadIdx.x] = l;
    }
    // ---- rescale O and accumulate O += P * V^T for this wave's (i,c) slice
#pragma unroll
    for (int t = 0; t < 8; ++t) {
#pragma unroll
      for (int e = 0; e < 8; ++e) acc[t][e] *= resc[is2 * 16 + mb + e];
    }
    for (int kk = 0; kk < 128; kk += 32) {
      v16h fa = load_frag(Ps + is2 * 16 * 128, 0, 128, kk, lane);  // A: rows=i,K=j
#pragma unroll
      for (int t = 0; t < 8; ++t) {
        v16h fb = load_frag(vb, cc * 128 + 16 * t, NPIX, j0 + kk, lane); // B: K=j,N=c
        acc[t] = wmma_f16(fa, fb, acc[t]);
      }
    }
    __syncthreads();
  }
  // ---- epilogue: normalize by row sums, write oT[n][c] f16
#pragma unroll
  for (int t = 0; t < 8; ++t) {
    int c = cc * 128 + 16 * t + (lane & 15);
#pragma unroll
    for (int e = 0; e < 8; ++e) {
      int r = is2 * 16 + mb + e;
      float o = acc[t][e] / rowl[r];
      oT[base + (size_t)(i0 + r) * CDIM + c] = (_Float16)o;
    }
  }
}

// ---------------- Output projection + bias + residual -------------------------
__global__ void out_kernel(const _Float16* __restrict__ oT, const _Float16* __restrict__ wo,
                           const float* __restrict__ bo, const float* __restrict__ x,
                           float* __restrict__ out) {
  int lane = threadIdx.x & 31;
  int w    = threadIdx.x >> 5;
  int m0 = blockIdx.y * 64 + (w & 3) * 16;
  int n0 = blockIdx.x * 32 + (w >> 2) * 16;
  int b  = blockIdx.z;
  const _Float16* ob = oT + (size_t)b * NPIX * CDIM;
  v8f acc = {};
  for (int k0 = 0; k0 < CDIM; k0 += 32) {
    v16h fb = load_frag(ob, n0, CDIM, k0, lane);
    v16h fa = load_frag(wo, m0, CDIM, k0, lane);
    acc = wmma_f16(fa, fb, acc);
  }
  int n  = n0 + (lane & 15);
  int mb = m0 + ((lane >> 4) << 3);
  size_t xb = (size_t)b * CDIM * NPIX;
#pragma unroll
  for (int e = 0; e < 8; ++e) {
    size_t idx = xb + (size_t)(mb + e) * NPIX + n;
    out[idx] = x[idx] + acc[e] + bo[mb + e];
  }
}

// ---------------- Host-side orchestration -------------------------------------
extern "C" void kernel_launch(void* const* d_in, const int* in_sizes, int n_in,
                              void* d_out, int out_size, void* d_ws, size_t ws_size,
                              hipStream_t stream) {
  (void)in_sizes; (void)n_in; (void)out_size; (void)ws_size;
  const float* x    = (const float*)d_in[0];
  const float* gn_w = (const float*)d_in[1];
  const float* gn_b = (const float*)d_in[2];
  const float* wq   = (const float*)d_in[3];
  const float* bq   = (const float*)d_in[4];
  const float* wk   = (const float*)d_in[5];
  const float* bk   = (const float*)d_in[6];
  const float* wv   = (const float*)d_in[7];
  const float* bv   = (const float*)d_in[8];
  const float* wo   = (const float*)d_in[9];
  const float* bo   = (const float*)d_in[10];
  float* out = (float*)d_out;

  char* ws = (char*)d_ws;
  const size_t WSZ = (size_t)CDIM * CDIM * sizeof(_Float16);         // 512 KB
  const size_t TSZ = (size_t)BATCH * NPIX * CDIM * sizeof(_Float16); // 8 MB
  float*    stats = (float*)ws;                 size_t off = 1024;
  _Float16* wq16  = (_Float16*)(ws + off);      off += WSZ;
  _Float16* wk16  = (_Float16*)(ws + off);      off += WSZ;
  _Float16* wv16  = (_Float16*)(ws + off);      off += WSZ;
  _Float16* wo16  = (_Float16*)(ws + off);      off += WSZ;
  _Float16* hnT   = (_Float16*)(ws + off);      off += TSZ;
  _Float16* qT    = (_Float16*)(ws + off);      off += TSZ;
  _Float16* kT    = (_Float16*)(ws + off);      off += TSZ;
  _Float16* vM    = (_Float16*)(ws + off);      off += TSZ;
  _Float16* oT    = (_Float16*)(ws + off);      off += TSZ;  // ~44 MB total

  gn_stats_kernel<<<BATCH * GROUPS, 256, 0, stream>>>(x, stats);
  w2h_kernel<<<(CDIM * CDIM + 255) / 256, 256, 0, stream>>>(wq, wk, wv, wo,
                                                            wq16, wk16, wv16, wo16);
  gn_apply_kernel<<<(BATCH * CDIM * NPIX) / 256, 256, 0, stream>>>(x, stats, gn_w, gn_b, hnT);
  qkv_kernel<<<dim3(NPIX / 32, CDIM / 64, BATCH), 256, 0, stream>>>(
      hnT, wq16, wk16, wv16, bq, bk, bv, qT, kT, vM);
  attn_kernel<<<dim3(NPIX / 32, BATCH), 256, 0, stream>>>(qT, kT, vM, oT);
  out_kernel<<<dim3(NPIX / 32, CDIM / 64, BATCH), 256, 0, stream>>>(oT, wo16, bo, x, out);
}
